// SotaAgent_25417616457953
// MI455X (gfx1250) — matrix operand order repaired and checked by reading, not compiled, not run on hardware
//
#include <hip/hip_runtime.h>
#include <hip/hip_bf16.h>

// ---------------------------------------------------------------------------
// MI455X (gfx1250) implementation.
// Compute-bound (~230 GFLOP vs ~80MB traffic) -> route all big GEMMs through
// v_wmma_f32_16x16x32_bf16 (fp32 inputs converted RNE to bf16 via the native
// hardware converter, f32 accumulate).
// GEMM: 128x128 block tile, 8 waves (4Mx2N), 32x64 per wave = 2x4 WMMA tiles,
// LDS double-buffered (2x16KB) so next-tile global loads overlap WMMA issue.
// Attention is algebraically reduced so K/V are never materialized.
// All GEMM launches here have M % 128 == 0 (M = 4096).
// ---------------------------------------------------------------------------

typedef __bf16 bf16_t;
typedef bf16_t bf16x2  __attribute__((ext_vector_type(2)));
typedef bf16_t bf16x4  __attribute__((ext_vector_type(4)));
typedef bf16_t bf16x8  __attribute__((ext_vector_type(8)));
typedef bf16_t bf16x16 __attribute__((ext_vector_type(16)));
typedef float  f32x8   __attribute__((ext_vector_type(8)));

// ---- stage one 128x32 A-tile + 32x128 B-tile into registers --------------
// GUARD=false: straight-line unguarded loads (interior tiles).
template <bool GUARD>
__device__ __forceinline__ void load_tiles(const float* __restrict__ A,
                                           const float* __restrict__ Bm,
                                           int N, int K, int lda, int ldb,
                                           int transB, int tileM, int tileN,
                                           int k0, int tid,
                                           float* ra, float* rb) {
    // A: per thread 4 chunks of 4 consecutive k  (128*32 / 256 = 16 elems)
#pragma unroll
    for (int it = 0; it < 4; ++it) {
        int idx = tid + it * 256;
        int m = idx >> 3, kc = (idx & 7) << 2;
        const float* p = A + (size_t)(tileM + m) * lda + (k0 + kc);
#pragma unroll
        for (int q = 0; q < 4; ++q) {
            if (GUARD)
                ra[it * 4 + q] = (k0 + kc + q < K) ? p[q] : 0.f;
            else
                ra[it * 4 + q] = p[q];
        }
    }
    // B: per thread 8 k-pairs for one n each (128*16 pairs / 256 = 8)
#pragma unroll
    for (int it = 0; it < 8; ++it) {
        int idx = tid + it * 256;
        int n = idx & 127, kp = idx >> 7;
        int gn = tileN + n, gk = k0 + kp * 2;
        float v0, v1;
        if (GUARD) {
            v0 = v1 = 0.f;
            if (gn < N) {
                if (transB) {
                    const float* p = Bm + (size_t)gn * ldb + gk;
                    if (gk < K)     v0 = p[0];
                    if (gk + 1 < K) v1 = p[1];
                } else {
                    if (gk < K)     v0 = Bm[(size_t)gk * ldb + gn];
                    if (gk + 1 < K) v1 = Bm[(size_t)(gk + 1) * ldb + gn];
                }
            }
        } else {
            if (transB) {
                const float* p = Bm + (size_t)gn * ldb + gk;
                v0 = p[0]; v1 = p[1];
            } else {
                v0 = Bm[(size_t)gk * ldb + gn];
                v1 = Bm[(size_t)(gk + 1) * ldb + gn];
            }
        }
        rb[it * 2]     = v0;
        rb[it * 2 + 1] = v1;
    }
}

// fp32 -> bf16 via native converter (fptrunc = RNE), packed vector LDS stores
__device__ __forceinline__ void store_tiles(bf16_t* sA, bf16_t* sB,
                                            int tid, const float* ra, const float* rb) {
#pragma unroll
    for (int it = 0; it < 4; ++it) {            // A: one b64 store per chunk
        int idx = tid + it * 256;
        int m = idx >> 3, kc = (idx & 7) << 2;
        bf16x4 p;
        p.x = (bf16_t)ra[it * 4 + 0];
        p.y = (bf16_t)ra[it * 4 + 1];
        p.z = (bf16_t)ra[it * 4 + 2];
        p.w = (bf16_t)ra[it * 4 + 3];
        *(bf16x4*)(sA + m * 32 + kc) = p;       // 8B aligned
    }
#pragma unroll
    for (int it = 0; it < 8; ++it) {            // B: one b32 store per pair
        int idx = tid + it * 256;
        int n = idx & 127, kp = idx >> 7;
        bf16x2 p;
        p.x = (bf16_t)rb[it * 2];
        p.y = (bf16_t)rb[it * 2 + 1];
        *(bf16x2*)(sB + n * 32 + kp * 2) = p;   // 4B aligned
    }
}

// ---------------------------------------------------------------------------
// C[M,N] = act(A[M,K] @ B[K,N] (+bias[N]) (+addend[M,N]));  M % 128 == 0.
// transB: B element [k][n] read from Bm[n*ldb + k] (torch-style W.T).
// ---------------------------------------------------------------------------
__global__ __launch_bounds__(256, 1)
void gemm_wmma_bf16(const float* __restrict__ A, const float* __restrict__ Bm,
                    const float* __restrict__ bias, const float* __restrict__ addend,
                    float* __restrict__ C,
                    int M, int N, int K, int lda, int ldb, int ldc,
                    int transB, int relu) {
    __shared__ __align__(16) bf16_t sA[2][128 * 32];  // [m][k]
    __shared__ __align__(16) bf16_t sB[2][128 * 32];  // [n][k] (B^T)

    const int tid  = threadIdx.x;
    const int lane = tid & 31;          // wave32
    const int wave = tid >> 5;
    const int wm   = (wave >> 1) * 32;  // wave M origin within tile
    const int wn   = (wave & 1) * 64;   // wave N origin within tile
    const int hi   = lane >> 4;         // lane half (K-half select)
    const int lo16 = lane & 15;
    const int tileM = blockIdx.y * 128;
    const int tileN = blockIdx.x * 128;
    const bool ntail = (tileN + 128) > N;

    f32x8 acc[2][4];
    const f32x8 zero = {0.f, 0.f, 0.f, 0.f, 0.f, 0.f, 0.f, 0.f};
#pragma unroll
    for (int i = 0; i < 2; ++i)
#pragma unroll
        for (int j = 0; j < 4; ++j) acc[i][j] = zero;

    const int KT = (K + 31) >> 5;
    float ra[16], rb[16];

    // ---- prime buffer 0 ----
    if (ntail || 32 > K)
        load_tiles<true >(A, Bm, N, K, lda, ldb, transB, tileM, tileN, 0, tid, ra, rb);
    else
        load_tiles<false>(A, Bm, N, K, lda, ldb, transB, tileM, tileN, 0, tid, ra, rb);
    store_tiles(sA[0], sB[0], tid, ra, rb);
    __syncthreads();

    for (int kt = 0; kt < KT; ++kt) {
        const int cur = kt & 1;
        const bool have_next = (kt + 1) < KT;

        // ---- issue next tile's global loads (overlap with WMMA below) ----
        if (have_next) {
            const int k0n = (kt + 1) << 5;
            if (ntail || (k0n + 32) > K)
                load_tiles<true >(A, Bm, N, K, lda, ldb, transB, tileM, tileN, k0n, tid, ra, rb);
            else
                load_tiles<false>(A, Bm, N, K, lda, ldb, transB, tileM, tileN, k0n, tid, ra, rb);
        }

        // ---- fragments (ISA 7.12.2 16-bit layouts, ds_load_b128 pairs) ----
        const bf16_t* cA = sA[cur];
        const bf16_t* cB = sB[cur];
        bf16x16 afrag[2], bfrag[4];
#pragma unroll
        for (int i = 0; i < 2; ++i) {
            int m = wm + i * 16 + lo16;        // A: lane holds row M=l%16
            bf16x8 alo = *(const bf16x8*)(cA + m * 32 + hi * 8);
            bf16x8 ahi = *(const bf16x8*)(cA + m * 32 + 16 + hi * 8);
            afrag[i] = __builtin_shufflevector(alo, ahi,
                0,1,2,3,4,5,6,7,8,9,10,11,12,13,14,15);
        }
#pragma unroll
        for (int j = 0; j < 4; ++j) {
            int n = wn + j * 16 + lo16;        // B: lane holds col N=l%16
            bf16x8 blo = *(const bf16x8*)(cB + n * 32 + hi * 8);
            bf16x8 bhi = *(const bf16x8*)(cB + n * 32 + 16 + hi * 8);
            bfrag[j] = __builtin_shufflevector(blo, bhi,
                0,1,2,3,4,5,6,7,8,9,10,11,12,13,14,15);
        }
        // ---- 8 WMMAs per K-step per wave ----
#pragma unroll
        for (int i = 0; i < 2; ++i)
#pragma unroll
            for (int j = 0; j < 4; ++j)
                acc[i][j] = __builtin_amdgcn_wmma_f32_16x16x32_bf16(
                    false, afrag[i], false, bfrag[j],
                    (short)0, acc[i][j], false, false);

        // ---- commit next tile to the alternate buffer ----
        if (have_next)
            store_tiles(sA[cur ^ 1], sB[cur ^ 1], tid, ra, rb);
        __syncthreads();
    }

    // ---- epilogue: lane l, VGPR r holds (M = r + 8*(l/16), N = l%16) ----
    // M % 128 == 0 for all launches -> only N needs guarding, and only in
    // the (uniform) N-tail block.
    if (!ntail) {
#pragma unroll
        for (int i = 0; i < 2; ++i)
#pragma unroll
            for (int j = 0; j < 4; ++j) {
                int nb = tileN + wn + j * 16 + lo16;
#pragma unroll
                for (int r = 0; r < 8; ++r) {
                    int mb = tileM + wm + i * 16 + r + hi * 8;
                    float v = acc[i][j][r];
                    if (bias)   v += bias[nb];
                    if (addend) v += addend[(size_t)mb * N + nb];
                    if (relu)   v = v > 0.f ? v : 0.f;
                    C[(size_t)mb * ldc + nb] = v;
                }
            }
    } else {
#pragma unroll
        for (int i = 0; i < 2; ++i)
#pragma unroll
            for (int j = 0; j < 4; ++j) {
                int nb = tileN + wn + j * 16 + lo16;
                if (nb < N) {
#pragma unroll
                    for (int r = 0; r < 8; ++r) {
                        int mb = tileM + wm + i * 16 + r + hi * 8;
                        float v = acc[i][j][r];
                        if (bias)   v += bias[nb];
                        if (addend) v += addend[(size_t)mb * N + nb];
                        if (relu)   v = v > 0.f ? v : 0.f;
                        C[(size_t)mb * ldc + nb] = v;
                    }
                }
            }
    }
}

// ---------------------------------------------------------------------------
// Attention without materializing K/V:
//   energy[b,n] = feats[n,b,:] . qd[b,:] / sqrt(128)   (bak/bek cancel in softmax)
//   mix[b,:]    = sum_n softmax(energy)[n] * feats[n,b,:]
// ---------------------------------------------------------------------------
__global__ void attn_mix_kernel(const float* __restrict__ feats,
                                const float* __restrict__ qd,
                                float* __restrict__ mix,
                                int Nag, int D, int Bsz) {
    __shared__ float sq[80];
    __shared__ float se[32];
    __shared__ float sw[32];
    const int b = blockIdx.x, tid = threadIdx.x;
    for (int d = tid; d < D; d += blockDim.x) sq[d] = qd[(size_t)b * D + d];
    __syncthreads();
    if (tid < Nag) {
        float e = 0.f;
        const float* f = feats + (size_t)tid * Bsz * D + (size_t)b * D;
        for (int d = 0; d < D; ++d) e += f[d] * sq[d];
        se[tid] = e * 0.08838834764831845f;  // 1/sqrt(A=128)
    }
    __syncthreads();
    float mx = -1e30f;
    for (int n = 0; n < Nag; ++n) mx = fmaxf(mx, se[n]);
    float den = 0.f;
    for (int n = 0; n < Nag; ++n) den += __expf(se[n] - mx);
    if (tid < Nag) sw[tid] = __expf(se[tid] - mx) / den;
    __syncthreads();
    for (int d = tid; d < D; d += blockDim.x) {
        float acc2 = 0.f;
        for (int n = 0; n < Nag; ++n)
            acc2 += sw[n] * feats[(size_t)n * Bsz * D + (size_t)b * D + d];
        mix[(size_t)b * D + d] = acc2;
    }
}

// GRU gate fusion (torch r,z,n order). hout = (1-z)*n + z*h_prev.
__global__ void gru_gates_kernel(const float* __restrict__ gi,
                                 const float* __restrict__ gh,
                                 const float* __restrict__ hprev,
                                 float* __restrict__ hout, int Bsz, int H) {
    int idx = blockIdx.x * blockDim.x + threadIdx.x;
    if (idx >= Bsz * H) return;
    int b = idx / H, j = idx - b * H;
    const float* gib = gi + (size_t)b * 3 * H;
    const float* ghb = gh + (size_t)b * 3 * H;
    float ir = gib[j], iz = gib[H + j], in_ = gib[2 * H + j];
    float hr = ghb[j], hz = ghb[H + j], hn  = ghb[2 * H + j];
    float r = 1.f / (1.f + __expf(-(ir + hr)));
    float z = 1.f / (1.f + __expf(-(iz + hz)));
    float n = tanhf(in_ + r * hn);
    hout[idx] = (1.f - z) * n + z * hprev[idx];
}

// wo_q: [B,H] @ [H,6] + b -> q[:, 0:6]
__global__ void woq_kernel(const float* __restrict__ h, const float* __restrict__ Wwo,
                           const float* __restrict__ bwo, float* __restrict__ qout,
                           int Bsz, int H, int NACT, int LDQ) {
    int idx = blockIdx.x * blockDim.x + threadIdx.x;
    if (idx >= Bsz * NACT) return;
    int b = idx / NACT, j = idx - b * NACT;
    float acc = bwo[j];
    const float* hb = h + (size_t)b * H;
    for (int k = 0; k < H; ++k) acc += hb[k] * Wwo[k * NACT + j];
    qout[(size_t)b * LDQ + j] = acc;
}

// aq column n: q[:, 6+n] = t[:,:] . Wa2 + ba2
__global__ void aq_kernel(const float* __restrict__ t, const float* __restrict__ Wa2,
                          const float* __restrict__ ba2, float* __restrict__ qout,
                          int Bsz, int H, int LDQ, int col) {
    int b = blockIdx.x * blockDim.x + threadIdx.x;
    if (b >= Bsz) return;
    float acc = ba2[0];
    const float* tb = t + (size_t)b * H;
    for (int k = 0; k < H; ++k) acc += tb[k] * Wa2[k];
    qout[(size_t)b * LDQ + col] = acc;
}

static inline void gemm(hipStream_t s, const float* A, const float* Bm,
                        const float* bias, const float* addend, float* C,
                        int M, int N, int K, int lda, int ldb, int ldc,
                        int transB, int relu) {
    dim3 grid((N + 127) / 128, (M + 127) / 128);
    gemm_wmma_bf16<<<grid, 256, 0, s>>>(A, Bm, bias, addend, C,
                                        M, N, K, lda, ldb, ldc, transB, relu);
}

extern "C" void kernel_launch(void* const* d_in, const int* in_sizes, int n_in,
                              void* d_out, int out_size, void* d_ws, size_t ws_size,
                              hipStream_t stream) {
    const int B = 4096, NA = 31, NE = 32, D_AL = 64, D_EN = 65;
    const int D_OWN = 128, A_ = 128, E = 256, H = 512, NACT = 6;

    const float* own  = (const float*)d_in[0];
    const float* af   = (const float*)d_in[1];   // [NA,B,64]
    const float* ef   = (const float*)d_in[2];   // [NE,B,65]
    const float* h0   = (const float*)d_in[3];
    const float* Wq   = (const float*)d_in[4];   const float* bq   = (const float*)d_in[5];
    const float* Wak  = (const float*)d_in[6];   /* bak (7): softmax-invariant */
    const float* Wav  = (const float*)d_in[8];   const float* bav  = (const float*)d_in[9];
    const float* Wek  = (const float*)d_in[10];  /* bek (11): softmax-invariant */
    const float* Wev  = (const float*)d_in[12];  const float* bev  = (const float*)d_in[13];
    const float* Wov  = (const float*)d_in[14];  const float* bov  = (const float*)d_in[15];
    const float* w_ih = (const float*)d_in[16];  const float* w_hh = (const float*)d_in[17];
    const float* b_ih = (const float*)d_in[18];  const float* b_hh = (const float*)d_in[19];
    const float* Wwo  = (const float*)d_in[20];  const float* bwo  = (const float*)d_in[21];
    const float* We1  = (const float*)d_in[22];  const float* be1  = (const float*)d_in[23];
    const float* We2  = (const float*)d_in[24];  const float* be2  = (const float*)d_in[25];
    const float* Wa1  = (const float*)d_in[26];  const float* ba1  = (const float*)d_in[27];
    const float* Wa2  = (const float*)d_in[28];  const float* ba2  = (const float*)d_in[29];

    float* qout = (float*)d_out;                         // [B, NACT+NE]
    float* hout = qout + (size_t)B * (NACT + NE);        // [B, H]

    // workspace layout (peak ~69 MB); phase-B buffers reuse the gi/gh region
    float* ws    = (float*)d_ws;
    float* query = ws;                                   // B*128
    float* qa    = query + (size_t)B * A_;               // B*64  (= Wak @ q)
    float* qe    = qa    + (size_t)B * D_AL;             // B*65  (= Wek @ q)
    float* afm   = qe    + (size_t)B * D_EN;             // B*64  (attn-mixed ally feats)
    float* efm   = afm   + (size_t)B * D_AL;             // B*65
    float* tot   = efm   + (size_t)B * D_EN;             // B*768
    float* gi    = tot   + (size_t)B * 3 * E;            // B*1536
    float* gh    = gi    + (size_t)B * 3 * H;            // B*1536
    float* hb    = gi;                                   // B*512 (reuse after gates)
    float* u     = gi + (size_t)B * H;                   // B*512
    float* ehn   = gi + (size_t)B * 2 * H;               // B*512
    float* tn    = gh;                                   // B*512

    // ---- projections ----
    gemm(stream, own,   Wq,  bq,  nullptr, query,   B, A_, D_OWN, D_OWN, A_, A_,    0, 0);
    gemm(stream, own,   Wov, bov, nullptr, tot,     B, E,  D_OWN, D_OWN, E,  3 * E, 0, 0);
    gemm(stream, query, Wak, nullptr, nullptr, qa,  B, D_AL, A_,  A_,  A_, D_AL,    1, 0);
    gemm(stream, query, Wek, nullptr, nullptr, qe,  B, D_EN, A_,  A_,  A_, D_EN,    1, 0);

    // ---- attention (feature mixing; K/V never materialized) ----
    attn_mix_kernel<<<B, 128, 0, stream>>>(af, qa, afm, NA, D_AL, B);
    attn_mix_kernel<<<B, 128, 0, stream>>>(ef, qe, efm, NE, D_EN, B);
    gemm(stream, afm, Wav, bav, nullptr, tot + E,     B, E, D_AL, D_AL, E, 3 * E, 0, 0);
    gemm(stream, efm, Wev, bev, nullptr, tot + 2 * E, B, E, D_EN, D_EN, E, 3 * E, 0, 0);

    // ---- GRU ----
    gemm(stream, tot, w_ih, b_ih, nullptr, gi, B, 3 * H, 3 * E, 3 * E, 3 * E, 3 * H, 1, 0);
    gemm(stream, h0,  w_hh, b_hh, nullptr, gh, B, 3 * H, H,     H,     H,     3 * H, 1, 0);
    gru_gates_kernel<<<(B * H + 255) / 256, 256, 0, stream>>>(gi, gh, h0, hout, B, H);

    // ---- action heads ----
    woq_kernel<<<(B * NACT + 255) / 256, 256, 0, stream>>>(hout, Wwo, bwo, qout,
                                                           B, H, NACT, NACT + NE);
    // hb = h @ Wa1[H:2H, :] + ba1  (hoisted out of the per-enemy loop)
    gemm(stream, hout, Wa1 + (size_t)H * H, ba1, nullptr, hb, B, H, H, H, H, H, 0, 0);

    for (int n = 0; n < NE; ++n) {
        const float* efn = ef + (size_t)n * B * D_EN;
        gemm(stream, efn, We1, be1, nullptr, u,   B, H, D_EN, D_EN, H, H, 0, 1); // relu
        gemm(stream, u,   We2, be2, nullptr, ehn, B, H, H,    H,    H, H, 0, 0);
        gemm(stream, ehn, Wa1, nullptr, hb, tn,   B, H, H,    H,    H, H, 0, 1); // relu(+hb)
        aq_kernel<<<(B + 255) / 256, 256, 0, stream>>>(tn, Wa2, ba2, qout,
                                                       B, H, NACT + NE, NACT + n);
    }
}
